// DEAM_with_Swsi_69561290326534
// MI455X (gfx1250) — matrix operand order repaired and checked by reading, not compiled, not run on hardware
//
#include <hip/hip_runtime.h>
#include <hip/hip_bf16.h>
#include <math.h>

typedef __attribute__((ext_vector_type(16))) _Float16 v16h;
typedef __attribute__((ext_vector_type(8)))  _Float16 v8h;
typedef __attribute__((ext_vector_type(8)))  float    v8f;
typedef __attribute__((ext_vector_type(4)))  unsigned int u32x4;
typedef __attribute__((ext_vector_type(8)))  int      i32x8;
typedef __attribute__((ext_vector_type(4)))  int      i32x4;

#define B_     2
#define C_     128
#define H_     512
#define W_     512
#define HP_    64
#define WP_    64
#define HWP_   4096
#define KC_    16
#define NSPLIT 4
#define MSEG   (HWP_ / NSPLIT)   // 1024 keys per split
#define NITER  (MSEG / 32)       // 32 key-chunks per wave

__device__ inline v8f wmma_f16(v16h a, v16h b, v8f c) {
  return __builtin_amdgcn_wmma_f32_16x16x32_f16(false, a, false, b, (short)0, c,
                                                false, false);
}

// 16-lane (row) max reduction with DPP16 — no LDS, no waits.
template <int CTRL>
__device__ __forceinline__ float dppmax(float x) {
  int y = __builtin_amdgcn_update_dpp(0, __float_as_int(x), CTRL, 0xf, 0xf, true);
  return fmaxf(x, __int_as_float(y));
}
__device__ __forceinline__ float rowmax16(float x) {
  x = dppmax<0xB1>(x);    // quad_perm [1,0,3,2]  (xor 1)
  x = dppmax<0x4E>(x);    // quad_perm [2,3,0,1]  (xor 2)
  x = dppmax<0x141>(x);   // row_half_mirror      (covers 8)
  x = dppmax<0x140>(x);   // row_mirror           (covers 16)
  return x;
}

// TDM: async DMA of one 2-D V chunk (tile 32 x 128 rows, f16, row stride 4096)
// from global [c][m] into a compact LDS tile [c][m_local] (8 KB).
// D# packing per CDNA5 ISA ch. 8 (group0: flags/lds/global/type, group1: dims).
__device__ __forceinline__ void tdm_load_vchunk(const _Float16* gsrc,
                                                unsigned lds_off) {
  unsigned long long ga = (unsigned long long)(size_t)gsrc;
  u32x4 g0;
  g0[0] = 1u;                                   // count=1 (valid), user mode
  g0[1] = lds_off;                              // LDS byte address
  g0[2] = (unsigned)ga;                         // global_addr[31:0]
  g0[3] = (unsigned)((ga >> 32) & 0x01FFFFFFull) | (2u << 30);  // [56:32]|type=2
  i32x8 g1;
  g1[0] = 0x00010000;                           // data_size=2B, no multicast
  g1[1] = (int)(4096u << 16);                   // tensor_dim0 = 4096
  g1[2] = (int)(128u << 16);                    // tensor_dim1 = 128 (c rows)
  g1[3] = (int)(32u << 16);                     // tile_dim0 = 32 (m)
  g1[4] = 128;                                  // tile_dim1 = 128, tile_dim2=0
  g1[5] = 4096;                                 // tensor_dim0_stride = 4096
  g1[6] = 0;
  g1[7] = 0;
  i32x4 g2 = (i32x4)0;                          // 2-D tensor: groups 2/3 unused
  i32x4 g3 = (i32x4)0;
  i32x8 g4 = (i32x8)0;                          // unused trailing group
  __builtin_amdgcn_tensor_load_to_lds(g0, g1, g2, g3, g4, 0);
}

// ---------------- pool 8x8 (input->xp, diff->dp), layout [b][pos][c] ---------
__global__ __launch_bounds__(256) void pool_kernel(
    const float* __restrict__ input, const float* __restrict__ diff,
    float* __restrict__ xp, float* __restrict__ dp) {
  unsigned tid = blockIdx.x * 256u + threadIdx.x;   // B*C*64*64 = 2^20
  int pw = tid & 63;
  int ph = (tid >> 6) & 63;
  int c  = (tid >> 12) & 127;
  int b  = tid >> 19;
  size_t base = (((size_t)(b * C_ + c) * H_) + (size_t)ph * 8) * W_ + (size_t)pw * 8;
  float sx = 0.f, sd = 0.f;
#pragma unroll
  for (int i = 0; i < 8; ++i) {
    const float* pi = input + base + (size_t)i * W_;
    const float* pd = diff  + base + (size_t)i * W_;
#pragma unroll
    for (int j = 0; j < 8; ++j) { sx += pi[j]; sd += pd[j]; }
  }
  int pos = ph * WP_ + pw;
  size_t o = ((size_t)b * HWP_ + pos) * C_ + c;
  xp[o] = sx * (1.f / 64.f);
  dp[o] = sd * (1.f / 64.f);
}

// ------------- bilinear(align_corners) 128x128 -> 64x64, sigmoid ------------
__global__ __launch_bounds__(256) void sig_kernel(
    const float* __restrict__ swsi, float* __restrict__ sigb) {
  unsigned tid = blockIdx.x * 256u + threadIdx.x;   // B*4096 = 8192
  int px = tid & 63;
  int py = (tid >> 6) & 63;
  int b  = tid >> 12;
  float fy = py * (127.0f / 63.0f);
  float fx = px * (127.0f / 63.0f);
  int y0 = (int)fy; float wy = fy - (float)y0; int y1 = min(y0 + 1, 127);
  int x0 = (int)fx; float wx = fx - (float)x0; int x1 = min(x0 + 1, 127);
  const float* sp = swsi + (size_t)b * 128 * 128;
  float r0 = sp[y0 * 128 + x0] * (1.f - wy) + sp[y1 * 128 + x0] * wy;
  float r1 = sp[y0 * 128 + x1] * (1.f - wy) + sp[y1 * 128 + x1] * wy;
  float v = r0 * (1.f - wx) + r1 * wx;
  sigb[(size_t)b * HWP_ + py * WP_ + px] = 1.f / (1.f + __expf(-v));
}

// ---------- q/k 1x1 conv projections -> f16 rows [b][pos][16] ----------------
__global__ __launch_bounds__(256) void qk_kernel(
    const float* __restrict__ dp, const float* __restrict__ Wq,
    const float* __restrict__ bq, const float* __restrict__ Wk,
    const float* __restrict__ bk, _Float16* __restrict__ q16,
    _Float16* __restrict__ k16) {
  unsigned tid = blockIdx.x * 256u + threadIdx.x;   // B*4096*16 = 131072
  int kc  = tid & 15;
  int pos = (tid >> 4) & 4095;
  int b   = tid >> 16;
  const float* drow = dp + ((size_t)b * HWP_ + pos) * C_;
  float aq = bq[kc], ak = bk[kc];
#pragma unroll 4
  for (int c = 0; c < C_; ++c) {
    float dv = drow[c];
    aq += dv * Wq[kc * C_ + c];
    ak += dv * Wk[kc * C_ + c];
  }
  size_t o = ((size_t)b * HWP_ + pos) * KC_ + kc;
  q16[o] = (_Float16)aq;
  k16[o] = (_Float16)ak;
}

// ------ v 1x1 conv + similarity gate -> f16 transposed [b][c][pos] ----------
__global__ __launch_bounds__(256) void v_kernel(
    const float* __restrict__ xp, const float* __restrict__ Wv,
    const float* __restrict__ bv, const float* __restrict__ sigb,
    _Float16* __restrict__ vgT) {
  unsigned tid = blockIdx.x * 256u + threadIdx.x;   // B*4096*128 = 2^20
  int o   = tid & 127;
  int pos = (tid >> 7) & 4095;
  int b   = tid >> 19;
  const float* xrow = xp + ((size_t)b * HWP_ + pos) * C_;
  float a = bv[o];
#pragma unroll 4
  for (int c = 0; c < C_; ++c) a += xrow[c] * Wv[o * C_ + c];
  float g = 1.0f + 0.1f * sigb[(size_t)b * HWP_ + pos];
  vgT[((size_t)(b * C_ + o)) * HWP_ + pos] = (_Float16)(a * g);
}

// ------------------- flash attention core (WMMA + TDM, split-K) -------------
// 8 waves/block share one (batch, key-segment); each wave owns a 16-query tile.
// V chunks are DMA'd to LDS by the Tensor Data Mover, double-buffered.
__global__ __launch_bounds__(256) void attn_kernel(
    const _Float16* __restrict__ q16, const _Float16* __restrict__ k16,
    const _Float16* __restrict__ vgT, const float* __restrict__ sigb,
    const float* __restrict__ beta_p, float* __restrict__ Opart,
    float* __restrict__ Mstat, float* __restrict__ Lstat) {
  __shared__ __align__(64) _Float16 pstage[8][16][32];
  __shared__ __align__(64) _Float16 vstage[2][C_ * 32];   // 2 x 8 KB V chunks

  const int wave    = threadIdx.x >> 5;
  const int lane    = threadIdx.x & 31;
  const int half_id = lane >> 4;     // 0: lanes 0-15, 1: lanes 16-31
  const int l16     = lane & 15;
  const int qt      = (blockIdx.x & 31) * 8 + wave;  // query tile (0..255)
  const int s       = (blockIdx.x >> 5) & 3;         // key segment
  const int b       = blockIdx.x >> 7;               // batch
  const int n0      = qt << 4;
  const int m0      = s * MSEG;
  const float beta  = *beta_p;

  const _Float16* vsrc0 = vgT + ((size_t)b * C_ * HWP_ + m0);
  const unsigned vls_off[2] = {(unsigned)(size_t)&vstage[0][0],
                               (unsigned)(size_t)&vstage[1][0]};

  // Q tile, A-matrix layout: lanes<16 hold K 0..15;
  // lanes>=16 (K 16..31) are the zero pad of the head dimension.
  v16h Aq;
  {
    const _Float16* qp = q16 + (((size_t)b * HWP_ + n0 + l16) * KC_);
    v16h qv = *(const v16h*)qp;
#pragma unroll
    for (int i = 0; i < 16; ++i) Aq[i] = half_id == 0 ? qv[i] : (_Float16)0.f;
  }

  v16h Bones;
#pragma unroll
  for (int i = 0; i < 16; ++i) Bones[i] = (_Float16)1.f;

  v8f M, L, O[8];
#pragma unroll
  for (int r = 0; r < 8; ++r) { M[r] = -1e30f; L[r] = 0.f; }
#pragma unroll
  for (int ct = 0; ct < 8; ++ct)
#pragma unroll
    for (int r = 0; r < 8; ++r) O[ct][r] = 0.f;

  v8f zc;
#pragma unroll
  for (int r = 0; r < 8; ++r) zc[r] = 0.f;

  // prologue: DMA first V chunk into buffer 0
  if (wave == 0) {
    tdm_load_vchunk(vsrc0, vls_off[0]);
    __builtin_amdgcn_s_wait_tensorcnt(0);
  }
  __syncthreads();

  for (int it = 0; it < NITER; ++it) {
    const int mb  = m0 + it * 32;
    const int cur = it & 1;

    // kick DMA for next chunk into the other buffer (last read one iter ago)
    if (wave == 0 && it + 1 < NITER)
      tdm_load_vchunk(vsrc0 + (it + 1) * 32, vls_off[cur ^ 1]);

    // ---- S = (Q Kt) for two 16-key subtiles --------------------------------
    v16h Bk;
#pragma unroll
    for (int i = 0; i < 16; ++i) Bk[i] = (_Float16)0.f;
    if (half_id == 0)   // lanes>=16 carry the zero-padded K range 16..31
      Bk = *(const v16h*)(k16 + ((size_t)b * HWP_ + mb + l16) * KC_);
    v8f S0 = wmma_f16(Aq, Bk, zc);

#pragma unroll
    for (int i = 0; i < 16; ++i) Bk[i] = (_Float16)0.f;
    if (half_id == 0)
      Bk = *(const v16h*)(k16 + ((size_t)b * HWP_ + mb + 16 + l16) * KC_);
    v8f S1 = wmma_f16(Aq, Bk, zc);

    // scale + key-position bias (column m = lane within half)
    float bias0 = beta * sigb[(size_t)b * HWP_ + mb + l16];
    float bias1 = beta * sigb[(size_t)b * HWP_ + mb + 16 + l16];
    v8f E0, E1;
#pragma unroll
    for (int r = 0; r < 8; ++r) {
      E0[r] = S0[r] * 0.25f + bias0;
      E1[r] = S1[r] * 0.25f + bias1;
    }

    // ---- online softmax max: pure-VALU DPP reduction inside each half ------
    v8f nM, alpha, P0, P1;
#pragma unroll
    for (int r = 0; r < 8; ++r) {
      float cm = rowmax16(fmaxf(E0[r], E1[r]));
      nM[r]    = fmaxf(M[r], cm);
      alpha[r] = __expf(M[r] - nM[r]);
      P0[r]    = __expf(E0[r] - nM[r]);
      P1[r]    = __expf(E1[r] - nM[r]);
      M[r]     = nM[r];
    }

    // ---- P: C-layout -> A-layout via LDS transpose -------------------------
#pragma unroll
    for (int r = 0; r < 8; ++r) {
      int row = r + 8 * half_id;
      pstage[wave][row][l16]      = (_Float16)P0[r];
      pstage[wave][row][16 + l16] = (_Float16)P1[r];
    }
    asm volatile("s_wait_dscnt 0x0" ::: "memory");   // wave-local LDS RAW

    v16h Ap;
    {
      const _Float16* pr = &pstage[wave][l16][0];
      v8h p0 = *(const v8h*)(pr + half_id * 16);       // K(m) 0..7 / 16..23
      v8h p1 = *(const v8h*)(pr + half_id * 16 + 8);   // K(m) 8..15 / 24..31
#pragma unroll
      for (int i = 0; i < 8; ++i) { Ap[i] = p0[i]; Ap[8 + i] = p1[i]; }
    }

    // ---- row-sum of (f16-rounded) P via WMMA against all-ones B ------------
    v8f rs = wmma_f16(Ap, Bones, zc);
#pragma unroll
    for (int r = 0; r < 8; ++r) L[r] = L[r] * alpha[r] + rs[r];

    // ---- O = alpha*O + P*Vg, V from the TDM-staged LDS chunk ---------------
#pragma unroll
    for (int ct = 0; ct < 8; ++ct)
#pragma unroll
      for (int r = 0; r < 8; ++r) O[ct][r] *= alpha[r];
    const _Float16* vls = &vstage[cur][0];
#pragma unroll
    for (int ct = 0; ct < 8; ++ct) {
      v16h Bv = *(const v16h*)(vls + ((ct * 16 + l16) << 5) + (half_id << 4));
      O[ct] = wmma_f16(Ap, Bv, O[ct]);
    }

    // drain this block's outstanding DMA, then hand buffers around
    if (wave == 0) __builtin_amdgcn_s_wait_tensorcnt(0);
    __syncthreads();
  }

  // ---- store unnormalized partials + per-row stats --------------------------
  const size_t sb = (size_t)(s * B_ + b);
#pragma unroll
  for (int ct = 0; ct < 8; ++ct) {
#pragma unroll
    for (int r = 0; r < 8; ++r) {
      int n = n0 + r + 8 * half_id;
      int c = ct * 16 + l16;
      Opart[(sb * C_ + c) * HWP_ + n] = O[ct][r];
    }
  }
  if (l16 == 0) {
#pragma unroll
    for (int r = 0; r < 8; ++r) {
      int n = n0 + r + 8 * half_id;
      Mstat[sb * HWP_ + n] = M[r];
      Lstat[sb * HWP_ + n] = L[r];
    }
  }
}

// -------- merge split-K partials: softmax renormalization across segments ---
__global__ __launch_bounds__(256) void combine_kernel(
    const float* __restrict__ Opart, const float* __restrict__ Mstat,
    const float* __restrict__ Lstat, float* __restrict__ attn) {
  unsigned tid = blockIdx.x * 256u + threadIdx.x;   // B*C*HWP = 2^20
  int n = tid & 4095;
  int c = (tid >> 12) & 127;
  int b = tid >> 19;
  float Mmax = -1e30f;
#pragma unroll
  for (int s = 0; s < NSPLIT; ++s)
    Mmax = fmaxf(Mmax, Mstat[(size_t)(s * B_ + b) * HWP_ + n]);
  float num = 0.f, den = 0.f;
#pragma unroll
  for (int s = 0; s < NSPLIT; ++s) {
    size_t sb = (size_t)(s * B_ + b);
    float w = __expf(Mstat[sb * HWP_ + n] - Mmax);
    den += w * Lstat[sb * HWP_ + n];
    num += w * Opart[(sb * C_ + c) * HWP_ + n];
  }
  attn[tid] = num / den;   // attn layout [b][c][n] == tid
}

// -------- bilinear(align_corners) 64->512 + gamma * out + input -------------
__global__ __launch_bounds__(256) void upsample_kernel(
    const float* __restrict__ attn, const float* __restrict__ input,
    const float* __restrict__ gamma_p, float* __restrict__ out) {
  unsigned tid = blockIdx.x * 256u + threadIdx.x;   // B*C*512*512 = 2^26
  int X  = tid & 511;
  int Y  = (tid >> 9) & 511;
  int bc = tid >> 18;
  float fy = Y * (63.0f / 511.0f);
  float fx = X * (63.0f / 511.0f);
  int y0 = (int)fy; float wy = fy - (float)y0; int y1 = min(y0 + 1, 63);
  int x0 = (int)fx; float wx = fx - (float)x0; int x1 = min(x0 + 1, 63);
  const float* base = attn + (size_t)bc * HWP_;
  float r0 = base[y0 * WP_ + x0] * (1.f - wy) + base[y1 * WP_ + x0] * wy;
  float r1 = base[y0 * WP_ + x1] * (1.f - wy) + base[y1 * WP_ + x1] * wy;
  float v = r0 * (1.f - wx) + r1 * wx;
  out[tid] = gamma_p[0] * v + input[tid];
}

extern "C" void kernel_launch(void* const* d_in, const int* in_sizes, int n_in,
                              void* d_out, int out_size, void* d_ws,
                              size_t ws_size, hipStream_t stream) {
  const float* input = (const float*)d_in[0];
  const float* diff  = (const float*)d_in[1];
  const float* swsi  = (const float*)d_in[2];
  const float* Wq    = (const float*)d_in[3];
  const float* bq    = (const float*)d_in[4];
  const float* Wk    = (const float*)d_in[5];
  const float* bk    = (const float*)d_in[6];
  const float* Wv    = (const float*)d_in[7];
  const float* bv    = (const float*)d_in[8];
  const float* beta  = (const float*)d_in[9];
  const float* gamma = (const float*)d_in[10];
  float* outp = (float*)d_out;

  char* ws = (char*)d_ws;
  const size_t sz_pool = (size_t)B_ * HWP_ * C_ * sizeof(float);      // 4 MB
  const size_t sz_qk   = (size_t)B_ * HWP_ * KC_ * sizeof(_Float16);  // 256 KB
  const size_t sz_v    = (size_t)B_ * C_ * HWP_ * sizeof(_Float16);   // 2 MB
  const size_t sz_sig  = (size_t)B_ * HWP_ * sizeof(float);           // 32 KB
  const size_t sz_attn = (size_t)B_ * C_ * HWP_ * sizeof(float);      // 4 MB
  const size_t sz_opar = (size_t)NSPLIT * B_ * C_ * HWP_ * sizeof(float); // 16 MB
  const size_t sz_stat = (size_t)NSPLIT * B_ * HWP_ * sizeof(float);  // 128 KB

  size_t off = 0;
  float*    dp    = (float*)(ws + off);    off += sz_pool;
  float*    xp    = (float*)(ws + off);    off += sz_pool;
  _Float16* q16   = (_Float16*)(ws + off); off += sz_qk;
  _Float16* k16   = (_Float16*)(ws + off); off += sz_qk;
  _Float16* vgT   = (_Float16*)(ws + off); off += sz_v;
  float*    sigb  = (float*)(ws + off);    off += sz_sig;
  float*    attn  = (float*)(ws + off);    off += sz_attn;
  float*    Opart = (float*)(ws + off);    off += sz_opar;
  float*    Mstat = (float*)(ws + off);    off += sz_stat;
  float*    Lstat = (float*)(ws + off);    off += sz_stat;

  pool_kernel<<<(B_ * C_ * HWP_) / 256, 256, 0, stream>>>(input, diff, xp, dp);
  sig_kernel<<<(B_ * HWP_) / 256, 256, 0, stream>>>(swsi, sigb);
  qk_kernel<<<(B_ * HWP_ * KC_) / 256, 256, 0, stream>>>(dp, Wq, bq, Wk, bk,
                                                         q16, k16);
  v_kernel<<<(B_ * HWP_ * C_) / 256, 256, 0, stream>>>(xp, Wv, bv, sigb, vgT);
  attn_kernel<<<(B_ * 256 * NSPLIT) / 8, 256, 0, stream>>>(
      q16, k16, vgT, sigb, beta, Opart, Mstat, Lstat);
  combine_kernel<<<(B_ * C_ * HWP_) / 256, 256, 0, stream>>>(Opart, Mstat,
                                                             Lstat, attn);
  upsample_kernel<<<(B_ * C_ * H_ * W_) / 256, 256, 0, stream>>>(attn, input,
                                                                 gamma, outp);
}
